// JustAnARLSTM_76355928588895
// MI455X (gfx1250) — matrix operand order repaired and checked by reading, not compiled, not run on hardware
//
#include <hip/hip_runtime.h>

#define HID   256
#define GATES 1024
#define KTOT  288      // 256 (h) + 32 (ct[2] + x[9] + pad)
#define KC    9        // 288 / 32 K-chunks
#define AST   296      // A_lds row stride in bf16 (592B, 16B-aligned)
#define BROWS 16       // batch rows per workgroup

typedef __bf16 bf16_t;
typedef __attribute__((ext_vector_type(16))) __bf16 v16bf;
typedef __attribute__((ext_vector_type(8)))  __bf16 v8bf;
typedef __attribute__((ext_vector_type(8)))  float  v8f;

// explicit global-address-space types so the opaque pointer still lowers to
// global_load_b128 (LOADcnt only) instead of flat_load_b128 (LOADcnt+DScnt)
typedef const __attribute__((address_space(1))) __bf16 gbf_t;
typedef const __attribute__((address_space(1))) v8bf   gv8bf_t;

union BF { v16bf v; v8bf h[2]; };

__device__ __forceinline__ bf16_t f2bf(float f) {
  unsigned u = __builtin_bit_cast(unsigned, f);
  unsigned r = (u + 0x7FFFu + ((u >> 16) & 1u)) >> 16;
  unsigned short s = (unsigned short)r;
  return __builtin_bit_cast(bf16_t, s);
}
__device__ __forceinline__ float fsig(float x) {
  float e = __expf(-x);
  return __builtin_amdgcn_rcpf(1.0f + e);
}
__device__ __forceinline__ float ftanh(float x) {
  float e = __expf(-2.0f * x);
  return (1.0f - e) * __builtin_amdgcn_rcpf(1.0f + e);
}

// load the 8 gate-tile B-fragments of K-chunk kc (16x global_load_b128)
__device__ __forceinline__ void loadB(BF (&b)[8], gbf_t* wlane, int kc) {
#pragma unroll
  for (int i = 0; i < 8; ++i) {
    gbf_t* wp = wlane + (size_t)i * 16 * KTOT + kc * 32;
    b[i].h[0] = *(gv8bf_t*)(wp);
    b[i].h[1] = *(gv8bf_t*)(wp + 8);
  }
}
// run the 8 WMMAs of one K-chunk
__device__ __forceinline__ void mmaChunk(v8f (&acc)[8], const v16bf& a,
                                         const BF (&b)[8], bool first) {
  if (first) {
    v8f z = {0.f, 0.f, 0.f, 0.f, 0.f, 0.f, 0.f, 0.f};
#pragma unroll
    for (int i = 0; i < 8; ++i)
      acc[i] = __builtin_amdgcn_wmma_f32_16x16x32_bf16(false, a, false, b[i].v,
                                                       (short)0, z, false, false);
  } else {
#pragma unroll
    for (int i = 0; i < 8; ++i)
      acc[i] = __builtin_amdgcn_wmma_f32_16x16x32_bf16(false, a, false, b[i].v,
                                                       (short)0, acc[i], false, false);
  }
}
__device__ __forceinline__ v16bf loadA(const bf16_t (&A_lds)[BROWS][AST],
                                       int l16, int half, int kc) {
  union { v16bf v; v8bf h[2]; } ua;
  ua.h[0] = *(const v8bf*)&A_lds[l16][kc * 32 + 16 * half];        // elems 0-7
  ua.h[1] = *(const v8bf*)&A_lds[l16][kc * 32 + 16 * half + 8];   // elems 8-15
  return ua.v;
}

// Build gate-permuted concatenated weights (bf16) + fused bias in workspace.
// Tile n (0..63) -> wave w=n/8 owns hidden cols [w*32, w*32+32); within a wave,
// tiles {0,1}=i, {2,3}=f, {4,5}=g, {6,7}=o for those same 32 hidden columns.
__global__ void arlstm_prep(const float* __restrict__ Wih, const float* __restrict__ Whh,
                            const float* __restrict__ bih, const float* __restrict__ bhh,
                            bf16_t* __restrict__ Wcat, float* __restrict__ bias) {
  int idx = blockIdx.x * blockDim.x + threadIdx.x;
  if (idx >= GATES * KTOT) return;
  int ng = idx / KTOT;
  int k  = idx % KTOT;
  int n    = ng >> 4;
  int rrow = ng & 15;
  int gt = (n & 7) >> 1;                    // gate type 0..3 (i,f,g,o)
  int hb = (n >> 3) * 32 + (n & 1) * 16;    // hidden column block
  int orow = gt * HID + hb + rrow;          // original row of W_ih/W_hh
  float v = 0.0f;
  if (k < HID)            v = Whh[orow * HID + k];
  else if (k < HID + 11)  v = Wih[orow * 11 + (k - HID)];
  Wcat[ng * KTOT + k] = f2bf(v);
  if (k == 0) bias[ng] = bih[orow] + bhh[orow];
}

// 8 waves on one WGP = exactly 2 waves/SIMD; declare that so the compiler's
// VGPR budget is ~512 (stops pressure-driven sinking of the prefetch loads).
__global__ __launch_bounds__(256, 2)
void arlstm_kernel(const float* __restrict__ init_state,
                   const float* __restrict__ xa,
                   const bf16_t* __restrict__ Wcat,
                   const float* __restrict__ bias,
                   const float* __restrict__ fcW,
                   const float* __restrict__ fcb,
                   const float* __restrict__ h0,
                   const float* __restrict__ c0,
                   float* __restrict__ m_out,
                   float* __restrict__ cfull,
                   int nT, int Txa) {
  __shared__ __align__(16) bf16_t A_lds[BROWS][AST];   // [h(256) | ct(2) | x(9) | 0-pad]
  __shared__ __align__(16) float  c_lds[BROWS][HID + 4];
  __shared__ __align__(16) float  fcw_lds[2][HID];

  const int tid  = threadIdx.x;
  const int lane = tid & 31;
  const int wv   = tid >> 5;        // wave 0..7, owns hidden cols [wv*32, wv*32+32)
  const int half = lane >> 4;
  const int l16  = lane & 15;
  const int r0   = blockIdx.x * BROWS;

  // ---------------- init ----------------
  for (int i = tid; i < 2 * HID; i += 256) fcw_lds[i / HID][i % HID] = fcW[i];
  for (int i = tid; i < BROWS * (AST - HID); i += 256) {
    int r = i / (AST - HID), q = i % (AST - HID);
    A_lds[r][HID + q] = f2bf(0.0f);
  }
  for (int i = tid; i < BROWS * HID; i += 256) {
    int r = i >> 8, j = i & 255;
    A_lds[r][j] = f2bf(h0[j]);
  }
  __syncthreads();
  if (tid < BROWS * 2) {
    int r = tid >> 1, p = tid & 1;
    float v = init_state[(r0 + r) * 2 + p];
    A_lds[r][HID + p] = f2bf(v);
    cfull[(size_t)(r0 + r) * (nT + 1) * 2 + p] = v;   // c_full[:,0,:]
  }
  // c state in registers: fragment layout matches WMMA C/D (lane l16=col, v+8*half=row)
  v8f cfrag[2];
#pragma unroll
  for (int p = 0; p < 2; ++p) {
    float cv = c0[wv * 32 + p * 16 + l16];
#pragma unroll
    for (int v = 0; v < 8; ++v) cfrag[p][v] = cv;
  }
  // per-lane loop-invariant small preloads (10 VGPRs)
  float bv[8];
#pragma unroll
  for (int i = 0; i < 8; ++i) bv[i] = bias[(wv * 8 + i) * 16 + l16];
  const float fcb0 = fcb[0], fcb1 = fcb[1];
  __syncthreads();

  // per-lane weight base: row (wv*8)*16 + l16, K-offset 16*half (B is K-sequential)
  unsigned long long wl =
      (unsigned long long)(Wcat + (size_t)((wv * 8) * 16 + l16) * KTOT + 16 * half);

  // double-buffered B fragments; prologue: chunk 0 of step 0
  BF bufA[8], bufB[8];
  loadB(bufA, (gbf_t*)wl, 0);

  // ---------------- time loop (serial recurrence) ----------------
  for (int t = 0; t < nT; ++t) {
    // Opaque-ify the weight pointer each iteration: stops LLVM hoisting the 72
    // loop-invariant weight-fragment loads out of the loop (scratch spills).
    asm volatile("" : "+v"(wl));
    gbf_t* wlane = (gbf_t*)wl;

    // stage x_t (ct already written by previous FC phase)
    if (tid < BROWS * 9) {
      int r = tid / 9, q = tid % 9;
      A_lds[r][HID + 2 + q] = f2bf(xa[((size_t)(r0 + r) * Txa + t) * 9 + q]);
    }
    __syncthreads();

    // ---- WMMA: K-chunk outer, software-pipelined double buffer ----
    // Pipeline pinned by sched_group_barrier per chunk:
    //   [2 ds reads: A(kc)] [16 vmem reads: B(kc+1)] [8 wmma: chunk kc]
    // so every chunk's wait tolerates 16 in-flight loads (in-order return).
    v8f acc[8];
#pragma unroll
    for (int kc = 0; kc < KC; ++kc) {
      if ((kc & 1) == 0) {
        if (kc + 1 < KC) loadB(bufB, wlane, kc + 1);
        v16bf a = loadA(A_lds, l16, half, kc);
        mmaChunk(acc, a, bufA, kc == 0);
      } else {
        loadB(bufA, wlane, kc + 1);     // kc odd -> kc+1 <= 8 always (KC=9)
        v16bf a = loadA(A_lds, l16, half, kc);
        mmaChunk(acc, a, bufB, false);
      }
      __builtin_amdgcn_sched_group_barrier(0x100, 2, 0);   // DS read: A(kc)
      if (kc + 1 < KC)
        __builtin_amdgcn_sched_group_barrier(0x020, 16, 0); // VMEM read: B(kc+1)
      __builtin_amdgcn_sched_group_barrier(0x008, 8, 0);   // WMMA: chunk kc
    }
    __syncthreads();   // all waves done reading A_lds before h overwrite

    // ---- LSTM elementwise, fully in registers (gate-matched tiles) ----
#pragma unroll
    for (int p = 0; p < 2; ++p) {
      v8f ig = acc[0 + p], fg = acc[2 + p], gg = acc[4 + p], og = acc[6 + p];
      int jc = wv * 32 + p * 16 + l16;
#pragma unroll
      for (int v = 0; v < 8; ++v) {
        int m = v + 8 * half;
        float iv = fsig(ig[v] + bv[0 + p]);
        float fv = fsig(fg[v] + bv[2 + p]);
        float gv = ftanh(gg[v] + bv[4 + p]);
        float ov = fsig(og[v] + bv[6 + p]);
        float cn = fv * cfrag[p][v] + iv * gv;
        float hn = ov * ftanh(cn);
        cfrag[p][v] = cn;
        c_lds[m][jc] = cn;              // for FC reduction
        A_lds[m][jc] = f2bf(hn);        // h broadcast for next step
        m_out[((size_t)(r0 + m) * nT + t) * HID + jc] = hn;
      }
    }
    // prefetch next step's chunk-0 B fragments: latency hides behind the
    // elementwise VALU work, the barriers and the FC phase
    loadB(bufA, wlane, 0);
    __syncthreads();   // c_lds / A_lds(h) complete

    // ---- FC: ct_new = c_new @ fcW.T + fcb  (32 tasks, 4 per wave) ----
#pragma unroll
    for (int q = 0; q < 4; ++q) {
      int task = wv * 4 + q;
      int r = task >> 1, p = task & 1;
      float s = 0.0f;
#pragma unroll
      for (int jj = 0; jj < HID / 32; ++jj) {
        int j = jj * 32 + lane;
        s += c_lds[r][j] * fcw_lds[p][j];
      }
#pragma unroll
      for (int m = 16; m >= 1; m >>= 1) s += __shfl_xor(s, m, 32);
      if (lane == 0) {
        float ct = s + (p ? fcb1 : fcb0);
        A_lds[r][HID + p] = f2bf(ct);   // feeds next step's lstm_in
        cfull[((size_t)(r0 + r) * (nT + 1) + (t + 1)) * 2 + p] = ct;
      }
    }
    // next iteration's top __syncthreads() orders ct/x writes vs WMMA reads
  }
}

extern "C" void kernel_launch(void* const* d_in, const int* in_sizes, int n_in,
                              void* d_out, int out_size, void* d_ws, size_t ws_size,
                              hipStream_t stream) {
  const float* init_state = (const float*)d_in[0];
  const float* xa   = (const float*)d_in[2];
  const float* W_ih = (const float*)d_in[3];
  const float* W_hh = (const float*)d_in[4];
  const float* b_ih = (const float*)d_in[5];
  const float* b_hh = (const float*)d_in[6];
  const float* fc_W = (const float*)d_in[7];
  const float* fc_b = (const float*)d_in[8];
  const float* h0   = (const float*)d_in[9];
  const float* c0   = (const float*)d_in[10];

  int B   = in_sizes[0] / 2;                                  // 256
  int Txa = in_sizes[2] / (B * 9);                            // xa time stride
  int nT  = (int)((((long long)out_size) / B - 2) / 258);     // out = B*(258*nT + 2)

  bf16_t* Wcat = (bf16_t*)d_ws;
  float*  bias = (float*)((char*)d_ws + (size_t)GATES * KTOT * sizeof(bf16_t));

  int total = GATES * KTOT;
  arlstm_prep<<<(total + 255) / 256, 256, 0, stream>>>(W_ih, W_hh, b_ih, b_hh, Wcat, bias);

  float* m_out = (float*)d_out;
  float* cfull = m_out + (size_t)B * nT * HID;

  arlstm_kernel<<<B / BROWS, 256, 0, stream>>>(init_state, xa, Wcat, bias, fc_W, fc_b,
                                               h0, c0, m_out, cfull, nT, Txa);
}
